// NeighborAttention_56762287784396
// MI455X (gfx1250) — matrix operand order repaired and checked by reading, not compiled, not run on hardware
//
#include <hip/hip_runtime.h>
#include <hip/hip_bf16.h>

// ---------------------------------------------------------------------------
// NeighborAttention for MI455X (gfx1250, wave32, WMMA, async global->LDS)
//   Q = X @ Wq              (bf16 WMMA GEMM)
//   K,V = hist @ Wk/Wv      (FUSED dual-B bf16 WMMA GEMM: hist read once)
//   9-neighbor scores/softmax/weighted-V (fp32 VALU, async LDS halo tiles)
//   out = ctx @ Wo          (bf16 WMMA GEMM, fp32 out)
// ---------------------------------------------------------------------------

typedef __bf16 bf16;
typedef __attribute__((ext_vector_type(16))) __bf16 v16bf;
typedef __attribute__((ext_vector_type(8)))  __bf16 v8bf;
typedef __attribute__((ext_vector_type(8)))  float  v8f;

union FragBF16 { v16bf v; v8bf h[2]; };

#define HW    128
#define DIM   512
#define NSTEP 6
#define NPIX  (HW * HW)

// ---- async global->LDS helpers (CDNA5 GLOBAL_LOAD_ASYNC_TO_LDS_B128) ------
// Probe-discovered signature: (v4i AS1*, v4i AS3*, imm offset, imm cpol)
typedef int v4i_vs __attribute__((vector_size(16)));
#define AS1P(p) ((__attribute__((address_space(1))) v4i_vs*)(p))
#define AS3P(p) ((__attribute__((address_space(3))) v4i_vs*)(p))

__device__ __forceinline__ void async_copy_16B(const void* g, void* l) {
#if __has_builtin(__builtin_amdgcn_global_load_async_to_lds_b128)
    __builtin_amdgcn_global_load_async_to_lds_b128(AS1P(g), AS3P(l), 0, 0);
#else
    *(int4*)l = *(const int4*)g;
#endif
}
__device__ __forceinline__ void async_wait_all() {
#if __has_builtin(__builtin_amdgcn_s_wait_asynccnt)
    __builtin_amdgcn_s_wait_asynccnt(0);
#elif __has_builtin(__builtin_amdgcn_global_load_async_to_lds_b128)
    asm volatile("s_wait_asynccnt 0x0" ::: "memory");
#endif
}
__device__ __forceinline__ void async_copy_32B(const void* g, void* l) {
    async_copy_16B(g, l);
    async_copy_16B((const char*)g + 16, (char*)l + 16);
}
__device__ __forceinline__ void async_copy_128B(const void* g, void* l) {
    #pragma unroll
    for (int i = 0; i < 8; ++i)
        async_copy_16B((const char*)g + i * 16, (char*)l + i * 16);
}

// ---------------- weight convert + transpose (fp32 [K][N] -> bf16 [N][K]) ---
__global__ void wconvT_kernel(const float* __restrict__ W, bf16* __restrict__ Wt) {
    int idx = blockIdx.x * 256 + threadIdx.x;      // 512*512 elements
    int k = idx >> 9;
    int n = idx & 511;
    Wt[(size_t)n * DIM + k] = (bf16)W[idx];
}

// ---- WMMA fragment loads (contiguous-K order, same for A and B), shared by both GEMMs --------
// A 16x32 bf16: lane holds row M=lane%16; 16 contiguous K (0..15 or 16..31)
// B 32x16 bf16: lane holds col N=lane%16; 16 contiguous K (0..15 or 16..31)
__device__ __forceinline__ void load_a_frag(FragBF16& a, const bf16* rowp, int ka) {
    a.h[0] = *(const v8bf*)(rowp + ka);
    a.h[1] = *(const v8bf*)(rowp + ka + 8);
}
__device__ __forceinline__ void load_b_frag(FragBF16& b, const bf16* rowp, int kb) {
    b.h[0] = *(const v8bf*)(rowp + kb);
    b.h[1] = *(const v8bf*)(rowp + kb + 8);
}

// ---------------- GEMM: C[M x 512] = A[M x 512] * B[512 x 512] --------------
// Bt is B transposed ([N][K], bf16). AF32: A fp32 (convert on load), else bf16
// staged via async-to-LDS. CF32: fp32 out, else bf16.
// Block = 256 thr (8 waves), tile 128x128, BK=32.
template<bool AF32, bool CF32>
__global__ __launch_bounds__(256)
void gemm512_kernel(const void* __restrict__ Ain, const bf16* __restrict__ Bt,
                    void* __restrict__ Cout) {
    __shared__ bf16 As[128][40];   // +8 bf16 pad to spread banks
    __shared__ bf16 Bs[128][40];   // [N][K] layout

    const int tid  = threadIdx.x;
    const int m0   = blockIdx.x * 128;
    const int n0   = blockIdx.y * 128;

    const int lane = tid & 31;
    const int wave = tid >> 5;
    const int wm   = (wave & 3) * 32;   // 4 waves across M
    const int wn   = (wave >> 2) * 64;  // 2 waves across N
    const int lr   = lane & 15;
    const bool hi  = lane >= 16;

    const v8f zero = {0.f, 0.f, 0.f, 0.f, 0.f, 0.f, 0.f, 0.f};
    v8f acc[2][4];
    #pragma unroll
    for (int im = 0; im < 2; ++im)
        #pragma unroll
        for (int in = 0; in < 4; ++in) acc[im][in] = zero;

    const int lm = tid >> 1;         // 0..127 : tile row (A: M, B: N)
    const int lk = (tid & 1) * 16;   // 0 or 16: K half

    for (int k0 = 0; k0 < DIM; k0 += 32) {
        __syncthreads();
        // ---- stage A tile ----
        if (AF32) {
            const float* ap = (const float*)Ain + (size_t)(m0 + lm) * DIM + k0 + lk;
            #pragma unroll
            for (int i = 0; i < 16; ++i) As[lm][lk + i] = (bf16)ap[i];
        } else {
            const bf16* ap = (const bf16*)Ain + (size_t)(m0 + lm) * DIM + k0 + lk;
            async_copy_32B(ap, &As[lm][lk]);
        }
        // ---- stage B tile (bf16 [N][K], direct async copy) ----
        async_copy_32B(Bt + (size_t)(n0 + lm) * DIM + k0 + lk, &Bs[lm][lk]);
        async_wait_all();
        __syncthreads();

        FragBF16 a[2], b[4];
        const int ka = hi ? 16 : 0;
        #pragma unroll
        for (int im = 0; im < 2; ++im)
            load_a_frag(a[im], &As[wm + im * 16 + lr][0], ka);
        const int kb = hi ? 16 : 0;
        #pragma unroll
        for (int in = 0; in < 4; ++in)
            load_b_frag(b[in], &Bs[wn + in * 16 + lr][0], kb);
        #pragma unroll
        for (int im = 0; im < 2; ++im)
            #pragma unroll
            for (int in = 0; in < 4; ++in)
                acc[im][in] = __builtin_amdgcn_wmma_f32_16x16x32_bf16(
                    false, a[im].v, false, b[in].v,
                    (short)0, acc[im][in], false, false);
    }

    // ---- store C: lane = col N=lane%16, VGPR r -> row r + 8*(lane>=16) ----
    const int rb = hi ? 8 : 0;
    #pragma unroll
    for (int im = 0; im < 2; ++im) {
        #pragma unroll
        for (int in = 0; in < 4; ++in) {
            const size_t row0 = (size_t)(m0 + wm + im * 16 + rb);
            const int    col  = n0 + wn + in * 16 + lr;
            #pragma unroll
            for (int r = 0; r < 8; ++r) {
                if (CF32)
                    ((float*)Cout)[(row0 + r) * DIM + col] = acc[im][in][r];
                else
                    ((bf16*)Cout)[(row0 + r) * DIM + col] = (bf16)acc[im][in][r];
            }
        }
    }
}

// ---------------- fused K+V projection: hist read ONCE ----------------------
// K = A @ Bt0^T, V = A @ Bt1^T ; A fp32, outputs bf16. 16 WMMAs per k-step.
__global__ __launch_bounds__(256)
void gemm512_dual_kernel(const float* __restrict__ Ain,
                         const bf16* __restrict__ Bt0, const bf16* __restrict__ Bt1,
                         bf16* __restrict__ C0, bf16* __restrict__ C1) {
    __shared__ bf16 As [128][40];
    __shared__ bf16 Bs0[128][40];
    __shared__ bf16 Bs1[128][40];

    const int tid  = threadIdx.x;
    const int m0   = blockIdx.x * 128;
    const int n0   = blockIdx.y * 128;

    const int lane = tid & 31;
    const int wave = tid >> 5;
    const int wm   = (wave & 3) * 32;
    const int wn   = (wave >> 2) * 64;
    const int lr   = lane & 15;
    const bool hi  = lane >= 16;

    const v8f zero = {0.f, 0.f, 0.f, 0.f, 0.f, 0.f, 0.f, 0.f};
    v8f acc[2][2][4];                       // [K/V][im][in] : 128 VGPRs
    #pragma unroll
    for (int s = 0; s < 2; ++s)
        #pragma unroll
        for (int im = 0; im < 2; ++im)
            #pragma unroll
            for (int in = 0; in < 4; ++in) acc[s][im][in] = zero;

    const int lm = tid >> 1;
    const int lk = (tid & 1) * 16;

    for (int k0 = 0; k0 < DIM; k0 += 32) {
        __syncthreads();
        {   // A tile: fp32 -> bf16 convert
            const float* ap = Ain + (size_t)(m0 + lm) * DIM + k0 + lk;
            #pragma unroll
            for (int i = 0; i < 16; ++i) As[lm][lk + i] = (bf16)ap[i];
        }
        async_copy_32B(Bt0 + (size_t)(n0 + lm) * DIM + k0 + lk, &Bs0[lm][lk]);
        async_copy_32B(Bt1 + (size_t)(n0 + lm) * DIM + k0 + lk, &Bs1[lm][lk]);
        async_wait_all();
        __syncthreads();

        FragBF16 a[2];
        const int ka = hi ? 16 : 0;
        #pragma unroll
        for (int im = 0; im < 2; ++im)
            load_a_frag(a[im], &As[wm + im * 16 + lr][0], ka);

        const int kb = hi ? 16 : 0;
        #pragma unroll
        for (int s = 0; s < 2; ++s) {       // reload B frags per matrix: caps VGPRs
            FragBF16 b[4];
            #pragma unroll
            for (int in = 0; in < 4; ++in)
                load_b_frag(b[in], s ? &Bs1[wn + in * 16 + lr][0]
                                     : &Bs0[wn + in * 16 + lr][0], kb);
            #pragma unroll
            for (int im = 0; im < 2; ++im)
                #pragma unroll
                for (int in = 0; in < 4; ++in)
                    acc[s][im][in] = __builtin_amdgcn_wmma_f32_16x16x32_bf16(
                        false, a[im].v, false, b[in].v,
                        (short)0, acc[s][im][in], false, false);
        }
    }

    const int rb = hi ? 8 : 0;
    #pragma unroll
    for (int s = 0; s < 2; ++s) {
        bf16* C = s ? C1 : C0;
        #pragma unroll
        for (int im = 0; im < 2; ++im)
            #pragma unroll
            for (int in = 0; in < 4; ++in) {
                const size_t row0 = (size_t)(m0 + wm + im * 16 + rb);
                const int    col  = n0 + wn + in * 16 + lr;
                #pragma unroll
                for (int r = 0; r < 8; ++r)
                    C[(row0 + r) * DIM + col] = (bf16)acc[s][im][in][r];
            }
    }
}

// ---------------- attention: 9-neighbor scores + softmax + weighted V ------
__device__ __forceinline__ void load_halo(bf16 (&sh)[18][18][64],
                                          const bf16* __restrict__ src,
                                          int t, int c, int by, int bx, int p) {
    int hy = p / 18, hx = p - hy * 18;
    int gy = min(max(by + hy - 1, 0), HW - 1);   // 'edge' padding
    int gx = min(max(bx + hx - 1, 0), HW - 1);
    async_copy_128B(src + (((size_t)t * NPIX + (size_t)gy * HW + gx) * DIM + c * 64),
                    &sh[hy][hx][0]);
}

__global__ __launch_bounds__(256)
void attn_kernel(const bf16* __restrict__ Qb, const bf16* __restrict__ Kb,
                 const bf16* __restrict__ Vb, bf16* __restrict__ Ctx) {
    __shared__ bf16 sh[18][18][64];            // 41.5 KB halo tile (K or V chunk)

    const int tid = threadIdx.x;
    const int ty = tid >> 4, tx = tid & 15;
    const int by = blockIdx.y * 16, bx = blockIdx.x * 16;
    const size_t pix = (size_t)(by + ty) * HW + (bx + tx);

    const int DY[9] = {-1,-1,-1, 0,0,0, 1,1,1};
    const int DX[9] = {-1, 0, 1,-1,0,1,-1,0,1};

    float sc[NSTEP][9];
    #pragma unroll
    for (int t = 0; t < NSTEP; ++t)
        #pragma unroll
        for (int n = 0; n < 9; ++n) sc[t][n] = 0.f;

    const int hp0 = tid, hp1 = tid + 256;       // 18*18 = 324 halo pixels

    // ---- pass 1: scores, D chunked by 64 ----
    for (int c = 0; c < DIM / 64; ++c) {
        float qf[64];
        const bf16* qp = Qb + pix * DIM + c * 64;
        #pragma unroll
        for (int d = 0; d < 64; ++d) qf[d] = (float)qp[d];

        #pragma unroll
        for (int t = 0; t < NSTEP; ++t) {
            __syncthreads();
            load_halo(sh, Kb, t, c, by, bx, hp0);
            if (hp1 < 324) load_halo(sh, Kb, t, c, by, bx, hp1);
            async_wait_all();
            __syncthreads();
            #pragma unroll
            for (int n = 0; n < 9; ++n) {
                const bf16* kp = &sh[ty + 1 + DY[n]][tx + 1 + DX[n]][0];
                float s = 0.f;
                #pragma unroll
                for (int d = 0; d < 64; ++d) s += qf[d] * (float)kp[d];
                sc[t][n] += s;
            }
        }
    }

    // ---- softmax over 9 neighbors, per frame ----
    const float scale = 0.044194173824159216f;  // 1/(sqrt(512)*TAU)
    float wts[NSTEP][9];
    #pragma unroll
    for (int t = 0; t < NSTEP; ++t) {
        float m = -3.4e38f;
        #pragma unroll
        for (int n = 0; n < 9; ++n) m = fmaxf(m, sc[t][n] * scale);
        float sum = 0.f;
        #pragma unroll
        for (int n = 0; n < 9; ++n) {
            float e = __expf(sc[t][n] * scale - m);
            wts[t][n] = e;
            sum += e;
        }
        const float inv = 1.f / sum;
        #pragma unroll
        for (int n = 0; n < 9; ++n) wts[t][n] *= inv;
    }

    // ---- pass 2: weighted V, mean over frames ----
    for (int c = 0; c < DIM / 64; ++c) {
        float acc2[64];
        #pragma unroll
        for (int d = 0; d < 64; ++d) acc2[d] = 0.f;

        #pragma unroll
        for (int t = 0; t < NSTEP; ++t) {
            __syncthreads();
            load_halo(sh, Vb, t, c, by, bx, hp0);
            if (hp1 < 324) load_halo(sh, Vb, t, c, by, bx, hp1);
            async_wait_all();
            __syncthreads();
            #pragma unroll
            for (int n = 0; n < 9; ++n) {
                const float w = wts[t][n];
                const bf16* vp = &sh[ty + 1 + DY[n]][tx + 1 + DX[n]][0];
                #pragma unroll
                for (int d = 0; d < 64; ++d) acc2[d] += w * (float)vp[d];
            }
        }
        bf16* op = Ctx + pix * DIM + c * 64;
        #pragma unroll
        for (int d = 0; d < 64; ++d) op[d] = (bf16)(acc2[d] * (1.0f / 6.0f));
    }
}

// ---------------------------------------------------------------------------
extern "C" void kernel_launch(void* const* d_in, const int* in_sizes, int n_in,
                              void* d_out, int out_size, void* d_ws, size_t ws_size,
                              hipStream_t stream) {
    const float* qfeat = (const float*)d_in[0];
    const float* hist8 = (const float*)d_in[1];
    const float* Wq    = (const float*)d_in[2];
    const float* Wk    = (const float*)d_in[3];
    const float* Wv    = (const float*)d_in[4];
    const float* Wo    = (const float*)d_in[5];
    // n_steps = min(T=8, LAYER_RANGE+1=6) -> last 6 frames
    const float* hist = hist8 + (size_t)2 * NPIX * DIM;

    char* ws = (char*)d_ws;
    size_t off = 0;
    const size_t WB = (size_t)DIM * DIM * sizeof(bf16);        // 512 KB each
    bf16* WqT = (bf16*)(ws + off); off += WB;
    bf16* WkT = (bf16*)(ws + off); off += WB;
    bf16* WvT = (bf16*)(ws + off); off += WB;
    bf16* WoT = (bf16*)(ws + off); off += WB;
    bf16* Qb  = (bf16*)(ws + off); off += (size_t)NPIX * DIM * sizeof(bf16);          // 16 MB
    bf16* Kb  = (bf16*)(ws + off); off += (size_t)NSTEP * NPIX * DIM * sizeof(bf16);  // 96 MB
    bf16* Vb  = (bf16*)(ws + off); off += (size_t)NSTEP * NPIX * DIM * sizeof(bf16);  // 96 MB
    bf16* Ctx = (bf16*)(ws + off); off += (size_t)NPIX * DIM * sizeof(bf16);          // 16 MB

    // 1) weights: fp32 [K][N] -> bf16 [N][K]
    wconvT_kernel<<<DIM * DIM / 256, 256, 0, stream>>>(Wq, WqT);
    wconvT_kernel<<<DIM * DIM / 256, 256, 0, stream>>>(Wk, WkT);
    wconvT_kernel<<<DIM * DIM / 256, 256, 0, stream>>>(Wv, WvT);
    wconvT_kernel<<<DIM * DIM / 256, 256, 0, stream>>>(Wo, WoT);

    // 2) projections: Q, then fused K+V (hist streamed once)
    gemm512_kernel<true, false><<<dim3(NPIX / 128, 4), 256, 0, stream>>>(qfeat, WqT, Qb);
    gemm512_dual_kernel<<<dim3(NSTEP * NPIX / 128, 4), 256, 0, stream>>>(hist, WkT, WvT, Kb, Vb);

    // 3) neighbor attention (fp32 VALU, async LDS halo tiles)
    attn_kernel<<<dim3(HW / 16, HW / 16), 256, 0, stream>>>(Qb, Kb, Vb, Ctx);

    // 4) output projection (fp32 result)
    gemm512_kernel<false, true><<<dim3(NPIX / 128, 4), 256, 0, stream>>>(Ctx, WoT, d_out);
}
